// SableNetwork_62749472194850
// MI455X (gfx1250) — matrix operand order repaired and checked by reading, not compiled, not run on hardware
//
#include <hip/hip_runtime.h>

// ---------------------------------------------------------------------------
// Types for CDNA5 WMMA (wave32, 16x16x32 bf16 -> f32) and TDM descriptors
// ---------------------------------------------------------------------------
typedef __bf16 bf16;
typedef __attribute__((ext_vector_type(8)))  bf16  v8bf;
typedef __attribute__((ext_vector_type(16))) bf16  v16bf;
typedef __attribute__((ext_vector_type(8)))  float v8f;
typedef __attribute__((ext_vector_type(4)))  unsigned int u32x4;
typedef __attribute__((ext_vector_type(8)))  int i32x8;
typedef __attribute__((ext_vector_type(4)))  int i32x4;

#define Bc   8
#define Sc   512
#define Ec   512
#define Hc   8
#define Ac   32
#define OBSc 128
#define NAc  8
#define DHc  64
#define Mc   (Bc*Sc)            /* 4096 rows */
#define MEc  ((size_t)Mc*Ec)    /* 2097152 elems */
#define Wsz  ((size_t)Ec*Ec)    /* 262144 elems per ExE weight */

__device__ __forceinline__ float gelu_f(float x){
  float x3 = x*x*x;
  return 0.5f*x*(1.0f + tanhf(0.7978845608028654f*(x + 0.044715f*x3)));
}
__device__ __forceinline__ float silu_f(float x){ return x/(1.0f + __expf(-x)); }

__device__ __forceinline__ v16bf cat16(v8bf a, v8bf b){
  return __builtin_shufflevector(a,b,0,1,2,3,4,5,6,7,8,9,10,11,12,13,14,15);
}
__device__ __forceinline__ v8f wmma_bf16(v16bf a, v16bf b, v8f c){
  return __builtin_amdgcn_wmma_f32_16x16x32_bf16(false, a, false, b, (short)0, c, false, false);
}

// ---------------------------------------------------------------------------
// Weight repack: f32 row-major (K x N) -> WMMA B-operand layout, bf16.
// packed[(((k/32)*(N/16) + n/16)*32 + lane)*16 + idx]
//   lane = (n%16) + 16*((k%32)/16), idx = k%16  (lanes 0-15: K low half)
// ---------------------------------------------------------------------------
__global__ void pack_b_kernel(const float* __restrict__ W, bf16* __restrict__ P,
                              int K, int N){
  int i = blockIdx.x*blockDim.x + threadIdx.x;
  if (i >= K*N) return;
  int k = i / N, n = i % N;
  int kt = k >> 5, nt = n >> 4;
  int lane = (n & 15) + (((k & 31) >> 4) << 4);
  int idx  = k & 15;
  P[ ((size_t)(kt*(N >> 4) + nt)*32 + lane)*16 + idx ] = (bf16)W[i];
}

// ---------------------------------------------------------------------------
// TDM-staged GEMM: C(MxN,f32) = A(MxK,bf16 row-major) @ Bpacked(KxN).
// Block (8 waves) owns a 128x64 macro-tile. Wave 0 issues one
// tensor_load_to_lds DMA of the K x 64-column packed-B panel (2D tile:
// (K/32) rows x 4KB, row stride (N/16)*1KB), waits TENSORcnt, block barriers,
// then each wave runs the K-loop: 1 A fetch feeds 4 WMMAs against LDS B.
// ---------------------------------------------------------------------------
__global__ void __launch_bounds__(256) gemm_tdm_kernel(
    const bf16* __restrict__ A, const bf16* __restrict__ Bp,
    float* __restrict__ C, int M, int N, int K, int lda)
{
  __shared__ bf16 Bpanel[512*64];                  // up to 64KB panel
  int lane = threadIdx.x & 31, wave = threadIdx.x >> 5;
  int lo = lane & 15, hi = lane >> 4, koff = hi*16;
  int ntn  = N >> 4;
  int nt0  = blockIdx.x*4;                         // 4 N-tiles = 64 cols
  int row0 = blockIdx.y*128 + wave*16;
  int nkt  = K >> 5;

#if __has_builtin(__builtin_amdgcn_tensor_load_to_lds)
  if (wave == 0){
    unsigned long long ga = (unsigned long long)(const void*)(Bp + (size_t)nt0*512);
    unsigned int lds = (unsigned int)(size_t)(void*)Bpanel;
    u32x4 g0 = {};
    g0[0] = 1u;                                          // count=1, user mode
    g0[1] = lds;                                         // lds_addr
    g0[2] = (unsigned int)(ga & 0xffffffffull);          // global_addr[31:0]
    g0[3] = (unsigned int)((ga >> 32) & 0x01ffffffull)   // global_addr[56:32]
          | 0x80000000u;                                 // type=2 (image)
    i32x8 g1 = {};
    g1[0] = (int)(1u << 16);                             // data_size = 2 bytes
    g1[1] = (int)(2048u << 16);                          // tensor_dim0[15:0]=2048
    g1[2] = (int)((unsigned)nkt << 16);                  // tensor_dim1[15:0]=K/32
    g1[3] = (int)(2048u << 16);                          // tile_dim0 = 2048 elems
    g1[4] = nkt;                                         // tile_dim1 = K/32
    g1[5] = ntn*512;                                     // dim0_stride (elems)
    g1[6] = 0; g1[7] = 0;
    i32x4 z4 = {};
    i32x8 z8 = {};
    __builtin_amdgcn_tensor_load_to_lds(g0, g1, z4, z4, z8, 0);
    __builtin_amdgcn_s_wait_tensorcnt(0);
  }
#else
  for (int i = threadIdx.x; i < nkt*2048; i += 256){
    int kt = i >> 11, e = i & 2047;
    Bpanel[i] = Bp[(size_t)(kt*ntn + nt0)*512 + e];
  }
#endif
  __syncthreads();

  const bf16* Ar = A + (size_t)(row0 + lo)*lda;
  v8f a0 = {}, a1 = {}, a2 = {}, a3 = {};
  for (int kt = 0; kt < nkt; ++kt){
    int k0 = kt << 5;
    v16bf a = cat16(*(const v8bf*)(Ar + k0 + koff),
                    *(const v8bf*)(Ar + k0 + 8 + koff));
    const bf16* Bl = Bpanel + kt*2048 + lane*16;
    a0 = wmma_bf16(a, *(const v16bf*)(Bl +    0), a0);
    a1 = wmma_bf16(a, *(const v16bf*)(Bl +  512), a1);
    a2 = wmma_bf16(a, *(const v16bf*)(Bl + 1024), a2);
    a3 = wmma_bf16(a, *(const v16bf*)(Bl + 1536), a3);
  }
  float* Cr = C + (size_t)(row0 + hi*8)*N + nt0*16 + lo;
#pragma unroll
  for (int r = 0; r < 8; ++r){
    Cr[(size_t)r*N +  0] = a0[r];
    Cr[(size_t)r*N + 16] = a1[r];
    Cr[(size_t)r*N + 32] = a2[r];
    Cr[(size_t)r*N + 48] = a3[r];
  }
}

// ---------------------------------------------------------------------------
// Simple per-wave GEMM (used for N=32 head projection)
// ---------------------------------------------------------------------------
__global__ void __launch_bounds__(256) gemm_bf16_kernel(
    const bf16* __restrict__ A, const bf16* __restrict__ Bp,
    float* __restrict__ C, int M, int N, int K, int lda)
{
  int lane = threadIdx.x & 31, wave = threadIdx.x >> 5;
  int ntn  = N >> 4;
  int tile = blockIdx.x*8 + wave;
  if (tile >= (M >> 4)*ntn) return;           // wave-uniform
  int mt = tile / ntn, nt = tile % ntn;
  int lo = lane & 15, hi = lane >> 4, koff = hi*16;

  const bf16* Ar = A + (size_t)(mt*16 + lo)*lda;
  const bf16* Bt = Bp + (size_t)nt*512 + (size_t)lane*16;
  size_t bstride = (size_t)ntn*512;

  v8f acc = {};
  for (int k0 = 0; k0 < K; k0 += 32){
    v16bf a = cat16(*(const v8bf*)(Ar + k0 + koff),
                    *(const v8bf*)(Ar + k0 + 8 + koff));
    v16bf b = *(const v16bf*)(Bt + (size_t)(k0 >> 5)*bstride);
    acc = wmma_bf16(a, b, acc);
  }
  float* Cr = C + (size_t)(mt*16 + hi*8)*N + nt*16 + lo;
#pragma unroll
  for (int r = 0; r < 8; ++r) Cr[(size_t)r*N] = acc[r];
}

// ---------------------------------------------------------------------------
// Flash-style retention core. One wave per (b, h, 16-row tile of n).
// Qb,Kb: (B,S,E) bf16 (Qb pre-scaled by 1/sqrt(DH)); VT: (B,H,DH,S) bf16.
// Decay gamma_h^(delta) * segmask applied analytically; no (S,S) matrix.
// ---------------------------------------------------------------------------
__global__ void __launch_bounds__(128) attention_kernel(
    const bf16* __restrict__ Qb, const bf16* __restrict__ Kb,
    const bf16* __restrict__ VT, const int* __restrict__ seg,
    float* __restrict__ Out, int causal)
{
  __shared__ bf16 sc[4][16][32];
  int lane = threadIdx.x & 31, w = threadIdx.x >> 5;
  int gw = blockIdx.x*4 + w;             // 0 .. B*H*(S/16)-1 = 2047
  int ntile = gw & 31;
  int h     = (gw >> 5) & 7;
  int b     = gw >> 8;
  int lo = lane & 15, hi = lane >> 4, koff = hi*16;

  const bf16* Qr  = Qb + ((size_t)(b*Sc + ntile*16 + lo))*Ec + h*DHc;
  const bf16* Kb0 = Kb + (size_t)b*Sc*Ec + h*DHc;
  const bf16* Vb0 = VT + ((size_t)(b*Hc + h))*DHc*Sc;
  const float lg  = log2f(1.0f - exp2f(-5.0f - (float)h));

  v16bf qa0 = cat16(*(const v8bf*)(Qr + koff),      *(const v8bf*)(Qr + 8 + koff));
  v16bf qa1 = cat16(*(const v8bf*)(Qr + 32 + koff), *(const v8bf*)(Qr + 40 + koff));

  int segn[8];
#pragma unroll
  for (int r = 0; r < 8; ++r) segn[r] = seg[b*Sc + ntile*16 + r + 8*hi];

  v8f o[4] = {};
  const int mend = causal ? (ntile*16 + 16) : Sc;
  for (int m0 = 0; m0 < mend; m0 += 32){
    const bf16* K0 = Kb0 + (size_t)(m0 + lo)*Ec;
    const bf16* K1 = K0 + (size_t)16*Ec;
    v8f s0 = {}, s1 = {};
    s0 = wmma_bf16(qa0, *(const v16bf*)(K0 + hi*16),      s0);
    s0 = wmma_bf16(qa1, *(const v16bf*)(K0 + 32 + hi*16), s0);
    s1 = wmma_bf16(qa0, *(const v16bf*)(K1 + hi*16),      s1);
    s1 = wmma_bf16(qa1, *(const v16bf*)(K1 + 32 + hi*16), s1);

    int sm0 = seg[b*Sc + m0 + lo];
    int sm1 = seg[b*Sc + m0 + 16 + lo];
#pragma unroll
    for (int r = 0; r < 8; ++r){
      int n  = ntile*16 + r + 8*hi;
      int d0 = n - (m0 + lo);
      int d1 = d0 - 16;
      float w0, w1;
      if (causal){
        w0 = (d0 >= 0 && segn[r] == sm0) ? exp2f(lg*(float)d0) : 0.0f;
        w1 = (d1 >= 0 && segn[r] == sm1) ? exp2f(lg*(float)d1) : 0.0f;
      } else {
        w0 = (segn[r] == sm0) ? exp2f(lg*fabsf((float)d0)) : 0.0f;
        w1 = (segn[r] == sm1) ? exp2f(lg*fabsf((float)d1)) : 0.0f;
      }
      sc[w][r + 8*hi][lo]      = (bf16)(s0[r]*w0);
      sc[w][r + 8*hi][16 + lo] = (bf16)(s1[r]*w1);
    }
    __builtin_amdgcn_wave_barrier();    // wave-internal LDS reshape (in-order DS)
    v16bf sa = cat16(*(const v8bf*)&sc[w][lo][koff],
                     *(const v8bf*)&sc[w][lo][8 + koff]);
    __builtin_amdgcn_wave_barrier();
#pragma unroll
    for (int dt = 0; dt < 4; ++dt){
      const bf16* Vr = Vb0 + (size_t)(dt*16 + lo)*Sc;
      v16bf bv = *(const v16bf*)(Vr + m0 + hi*16);
      o[dt] = wmma_bf16(sa, bv, o[dt]);
    }
  }
  float* Or = Out + ((size_t)(b*Sc + ntile*16 + 8*hi))*Ec + h*DHc + lo;
#pragma unroll
  for (int dt = 0; dt < 4; ++dt)
#pragma unroll
    for (int r = 0; r < 8; ++r) Or[(size_t)r*Ec + dt*16] = o[dt][r];
}

// ---------------------------------------------------------------------------
// Elementwise / row kernels
// ---------------------------------------------------------------------------
__global__ void __launch_bounds__(256) gelu_rmsnorm_kernel(
    const float* __restrict__ in, const float* __restrict__ gamma,
    float* __restrict__ outf, bf16* __restrict__ outb)
{
  __shared__ float red[256];
  int row = blockIdx.x, t = threadIdx.x;
  size_t base = (size_t)row*Ec;
  float x0 = gelu_f(in[base + t]);
  float x1 = gelu_f(in[base + t + 256]);
  red[t] = x0*x0 + x1*x1;
  __syncthreads();
  for (int off = 128; off > 0; off >>= 1){
    if (t < off) red[t] += red[t + off];
    __syncthreads();
  }
  float inv = rsqrtf(red[0]*(1.0f/(float)Ec) + 1e-6f);
  float y0 = x0*gamma[t]*inv, y1 = x1*gamma[t+256]*inv;
  outf[base+t] = y0;       outf[base+t+256] = y1;
  outb[base+t] = (bf16)y0; outb[base+t+256] = (bf16)y1;
}

__global__ void __launch_bounds__(256) add_rmsnorm_kernel(
    const float* __restrict__ a, const float* __restrict__ bsrc,
    const float* __restrict__ gamma, float* __restrict__ outf,
    bf16* __restrict__ outb)
{
  __shared__ float red[256];
  int row = blockIdx.x, t = threadIdx.x;
  size_t base = (size_t)row*Ec;
  float x0 = a[base + t]       + bsrc[base + t];
  float x1 = a[base + t + 256] + bsrc[base + t + 256];
  red[t] = x0*x0 + x1*x1;
  __syncthreads();
  for (int off = 128; off > 0; off >>= 1){
    if (t < off) red[t] += red[t + off];
    __syncthreads();
  }
  float inv = rsqrtf(red[0]*(1.0f/(float)Ec) + 1e-6f);
  float y0 = x0*gamma[t]*inv, y1 = x1*gamma[t+256]*inv;
  outf[base+t] = y0;       outf[base+t+256] = y1;
  outb[base+t] = (bf16)y0; outb[base+t+256] = (bf16)y1;
}

__global__ void groupnorm_kernel(const float* __restrict__ in, float* __restrict__ out){
  int g = blockIdx.x*blockDim.x + threadIdx.x;      // (B*S*H) groups of DH
  if (g >= Mc*Hc) return;
  const float* p = in + (size_t)g*DHc;
  float mu = 0.0f;
  for (int i = 0; i < DHc; ++i) mu += p[i];
  mu *= (1.0f/(float)DHc);
  float var = 0.0f;
  for (int i = 0; i < DHc; ++i){ float d = p[i]-mu; var += d*d; }
  var *= (1.0f/(float)DHc);
  float inv = rsqrtf(var + 1e-6f);
  float* q = out + (size_t)g*DHc;
  for (int i = 0; i < DHc; ++i) q[i] = (p[i]-mu)*inv;
}

__global__ void silu_mul_kernel(const float* __restrict__ g, const float* __restrict__ u,
                                bf16* __restrict__ out){
  size_t i = (size_t)blockIdx.x*blockDim.x + threadIdx.x;
  if (i >= MEc) return;
  out[i] = (bf16)(silu_f(g[i])*u[i]);
}

__global__ void cvt_kernel(const float* __restrict__ in, bf16* __restrict__ out,
                           float scale, int n){
  int i = blockIdx.x*blockDim.x + threadIdx.x;
  if (i >= n) return;
  out[i] = (bf16)(in[i]*scale);
}

__global__ void transpose_v_kernel(const float* __restrict__ in, bf16* __restrict__ out){
  int i = blockIdx.x*blockDim.x + threadIdx.x;
  if (i >= (int)MEc) return;
  int c = i & 511, rs = i >> 9;
  int s = rs & 511, b = rs >> 9;
  int h = c >> 6, d = c & 63;
  out[ (((size_t)(b*Hc + h)*DHc + d) << 9) + s ] = (bf16)in[i];
}

__global__ void embed_kernel(const int* __restrict__ action, const float* __restrict__ act_w,
                             float* __restrict__ out){
  int i = blockIdx.x*blockDim.x + threadIdx.x;
  if (i >= (int)MEc) return;
  int e = i & 511, rs = i >> 9;
  int s = rs & 511, b = rs >> 9;
  int a = ((s & (NAc-1)) == 0) ? 0 : (action[b*Sc + s - 1] + 1);
  out[i] = act_w[(size_t)a*Ec + e];
}

__global__ void seg_kernel(const unsigned char* __restrict__ dones, int* __restrict__ seg){
  int b = threadIdx.x;
  if (b >= Bc) return;
  int run = 0;
  seg[b*Sc] = 0;
  for (int s = 1; s < Sc; ++s){
    run += dones[b*Sc + s - 1] ? 1 : 0;
    seg[b*Sc + s] = run;
  }
}

__global__ void mask_kernel(const float* __restrict__ logits,
                            const unsigned char* __restrict__ legal,
                            float* __restrict__ out, int n){
  int i = blockIdx.x*blockDim.x + threadIdx.x;
  if (i >= n) return;
  out[i] = legal[i] ? logits[i] : -3.4028234663852886e38f;
}

// ---------------------------------------------------------------------------
// Host-side helpers (all launches on `stream`, graph-capture safe)
// ---------------------------------------------------------------------------
static inline void h_gemm_tdm(const bf16* A, const bf16* Bp, float* C,
                              int M, int N, int K, int lda, hipStream_t s){
  gemm_tdm_kernel<<<dim3(N/64, M/128), dim3(256), 0, s>>>(A, Bp, C, M, N, K, lda);
}
static inline void h_gemm_small(const bf16* A, const bf16* Bp, float* C,
                                int M, int N, int K, int lda, hipStream_t s){
  int tiles = (M >> 4)*(N >> 4);
  gemm_bf16_kernel<<<dim3((tiles + 7) >> 3), dim3(256), 0, s>>>(A, Bp, C, M, N, K, lda);
}
static inline void h_pack(const float* W, bf16* P, int K, int N, hipStream_t s){
  int n = K*N;
  pack_b_kernel<<<dim3((n + 255)/256), dim3(256), 0, s>>>(W, P, K, N);
}

struct Temps {
  float *Qf, *Kf, *Vf, *Gf, *ATT, *GN, *T;
  bf16  *Qb, *Kb, *VTb, *Gb;
  int   *seg;
};

static void run_retention(const float* qin_f, const bf16* qin_b, const bf16* kv_b,
                          const bf16* pk, int causal, const float* ln_gamma,
                          float* out_f, bf16* out_b, const Temps& t, hipStream_t s){
  h_gemm_tdm(qin_b, pk + 0*Wsz, t.Qf, Mc, Ec, Ec, Ec, s);
  h_gemm_tdm(kv_b,  pk + 1*Wsz, t.Kf, Mc, Ec, Ec, Ec, s);
  h_gemm_tdm(kv_b,  pk + 2*Wsz, t.Vf, Mc, Ec, Ec, Ec, s);
  h_gemm_tdm(qin_b, pk + 3*Wsz, t.Gf, Mc, Ec, Ec, Ec, s);
  cvt_kernel<<<dim3(MEc/256), dim3(256), 0, s>>>(t.Qf, t.Qb, 0.125f, (int)MEc); // 1/sqrt(DH)
  cvt_kernel<<<dim3(MEc/256), dim3(256), 0, s>>>(t.Kf, t.Kb, 1.0f,   (int)MEc);
  transpose_v_kernel<<<dim3(MEc/256), dim3(256), 0, s>>>(t.Vf, t.VTb);
  attention_kernel<<<dim3(512), dim3(128), 0, s>>>(t.Qb, t.Kb, t.VTb, t.seg, t.ATT, causal);
  groupnorm_kernel<<<dim3((Mc*Hc + 255)/256), dim3(256), 0, s>>>(t.ATT, t.GN);
  silu_mul_kernel<<<dim3(MEc/256), dim3(256), 0, s>>>(t.Gf, t.GN, t.Gb);
  h_gemm_tdm(t.Gb, pk + 4*Wsz, t.T, Mc, Ec, Ec, Ec, s);
  add_rmsnorm_kernel<<<dim3(Mc), dim3(256), 0, s>>>(qin_f, t.T, ln_gamma, out_f, out_b);
}

static void run_swiglu(const float* in_f, const bf16* in_b, const bf16* pk,
                       const float* ln_gamma, float* out_f, bf16* out_b,
                       const Temps& t, hipStream_t s){
  h_gemm_tdm(in_b, pk + 0*Wsz, t.Qf, Mc, Ec, Ec, Ec, s);   // gate
  h_gemm_tdm(in_b, pk + 1*Wsz, t.Kf, Mc, Ec, Ec, Ec, s);   // up
  silu_mul_kernel<<<dim3(MEc/256), dim3(256), 0, s>>>(t.Qf, t.Kf, t.Gb);
  h_gemm_tdm(t.Gb, pk + 2*Wsz, t.T, Mc, Ec, Ec, Ec, s);    // down
  add_rmsnorm_kernel<<<dim3(Mc), dim3(256), 0, s>>>(in_f, t.T, ln_gamma, out_f, out_b);
}

extern "C" void kernel_launch(void* const* d_in, const int* in_sizes, int n_in,
                              void* d_out, int out_size, void* d_ws, size_t ws_size,
                              hipStream_t stream)
{
  (void)in_sizes; (void)n_in; (void)out_size; (void)ws_size;
  const float* obs      = (const float*)d_in[0];
  const int*   action   = (const int*)d_in[1];
  const unsigned char* legal = (const unsigned char*)d_in[2];
  const unsigned char* dones = (const unsigned char*)d_in[3];
  const float* obs_w    = (const float*)d_in[4];
  const float* enc_ln0  = (const float*)d_in[5];
  const float* enc_ret  = (const float*)d_in[6];
  const float* enc_ln   = (const float*)d_in[7];
  const float* enc_ffn  = (const float*)d_in[8];
  const float* act_w    = (const float*)d_in[9];
  const float* dec_ln0  = (const float*)d_in[10];
  const float* dec_ret1 = (const float*)d_in[11];
  const float* dec_ret2 = (const float*)d_in[12];
  const float* dec_ln   = (const float*)d_in[13];
  const float* dec_ffn  = (const float*)d_in[14];
  const float* head_w1  = (const float*)d_in[15];
  const float* head_ln  = (const float*)d_in[16];
  const float* head_w2  = (const float*)d_in[17];

  // ---- workspace bump allocator ----
  char* p = (char*)d_ws;
  auto alloc = [&](size_t bytes)->char*{
    char* r = p; p += (bytes + 255) & ~(size_t)255; return r;
  };
  float* X   = (float*)alloc(MEc*4);   // encoder stream / obs_rep
  float* Y   = (float*)alloc(MEc*4);   // decoder stream
  float* Y2  = (float*)alloc(MEc*4);
  Temps t;
  t.Qf  = (float*)alloc(MEc*4);
  t.Kf  = (float*)alloc(MEc*4);
  t.Vf  = (float*)alloc(MEc*4);
  t.Gf  = (float*)alloc(MEc*4);
  t.ATT = (float*)alloc(MEc*4);
  t.GN  = (float*)alloc(MEc*4);
  t.T   = (float*)alloc(MEc*4);
  bf16* Xb  = (bf16*)alloc(MEc*2);
  bf16* Yb  = (bf16*)alloc(MEc*2);
  bf16* Y2b = (bf16*)alloc(MEc*2);
  t.Qb  = (bf16*)alloc(MEc*2);
  t.Kb  = (bf16*)alloc(MEc*2);
  t.VTb = (bf16*)alloc(MEc*2);
  t.Gb  = (bf16*)alloc(MEc*2);
  bf16* ObsB = (bf16*)alloc((size_t)Mc*OBSc*2);
  t.seg = (int*)alloc((size_t)Mc*4);

  // packed weights (bf16, WMMA B layout)
  bf16* PK = (bf16*)alloc((size_t)11354112*2);
  bf16* pk_obs  = PK;
  bf16* pk_eret = pk_obs  + (size_t)OBSc*Ec;   // 10 * Wsz
  bf16* pk_effn = pk_eret + 10*Wsz;            // 6 * Wsz
  bf16* pk_d1   = pk_effn + 6*Wsz;             // 10 * Wsz
  bf16* pk_d2   = pk_d1   + 10*Wsz;            // 10 * Wsz
  bf16* pk_dffn = pk_d2   + 10*Wsz;            // 6 * Wsz
  bf16* pk_hw1  = pk_dffn + 6*Wsz;             // 1 * Wsz
  bf16* pk_hw2  = pk_hw1  + Wsz;               // 512*32

  // ---- repack all weights (deterministic, every call) ----
  h_pack(obs_w, pk_obs, OBSc, Ec, stream);
  for (int i = 0; i < 10; ++i) h_pack(enc_ret  + (size_t)i*Wsz, pk_eret + (size_t)i*Wsz, Ec, Ec, stream);
  for (int i = 0; i < 6;  ++i) h_pack(enc_ffn  + (size_t)i*Wsz, pk_effn + (size_t)i*Wsz, Ec, Ec, stream);
  for (int i = 0; i < 10; ++i) h_pack(dec_ret1 + (size_t)i*Wsz, pk_d1   + (size_t)i*Wsz, Ec, Ec, stream);
  for (int i = 0; i < 10; ++i) h_pack(dec_ret2 + (size_t)i*Wsz, pk_d2   + (size_t)i*Wsz, Ec, Ec, stream);
  for (int i = 0; i < 6;  ++i) h_pack(dec_ffn  + (size_t)i*Wsz, pk_dffn + (size_t)i*Wsz, Ec, Ec, stream);
  h_pack(head_w1, pk_hw1, Ec, Ec, stream);
  h_pack(head_w2, pk_hw2, Ec, Ac, stream);

  // segment ids from dones
  seg_kernel<<<dim3(1), dim3(32), 0, stream>>>(dones, t.seg);

  // ---- encoder ----
  cvt_kernel<<<dim3((Mc*OBSc)/256), dim3(256), 0, stream>>>(obs, ObsB, 1.0f, Mc*OBSc);
  h_gemm_tdm(ObsB, pk_obs, t.T, Mc, Ec, OBSc, OBSc, stream);
  gelu_rmsnorm_kernel<<<dim3(Mc), dim3(256), 0, stream>>>(t.T, enc_ln0, X, Xb);
  for (int i = 0; i < 2; ++i){
    run_retention(X, Xb, Xb, pk_eret + (size_t)i*5*Wsz, /*causal=*/0,
                  enc_ln + (size_t)(i*2 + 0)*Ec, X, Xb, t, stream);
    run_swiglu(X, Xb, pk_effn + (size_t)i*3*Wsz,
               enc_ln + (size_t)(i*2 + 1)*Ec, X, Xb, t, stream);
  }
  // X / Xb now hold obs_rep

  // ---- decoder ----
  embed_kernel<<<dim3(MEc/256), dim3(256), 0, stream>>>(action, act_w, t.T);
  gelu_rmsnorm_kernel<<<dim3(Mc), dim3(256), 0, stream>>>(t.T, dec_ln0, Y, Yb);
  for (int i = 0; i < 2; ++i){
    run_retention(Y, Yb, Yb, pk_d1 + (size_t)i*5*Wsz, /*causal=*/1,
                  dec_ln + (size_t)(i*3 + 0)*Ec, Y, Yb, t, stream);
    run_retention(X, Xb, Yb, pk_d2 + (size_t)i*5*Wsz, /*causal=*/1,
                  dec_ln + (size_t)(i*3 + 1)*Ec, Y2, Y2b, t, stream);
    run_swiglu(Y2, Y2b, pk_dffn + (size_t)i*3*Wsz,
               dec_ln + (size_t)(i*3 + 2)*Ec, Y, Yb, t, stream);
  }

  // ---- head ----
  h_gemm_tdm(Yb, pk_hw1, t.T, Mc, Ec, Ec, Ec, stream);
  gelu_rmsnorm_kernel<<<dim3(Mc), dim3(256), 0, stream>>>(t.T, head_ln, t.GN, t.Gb);
  h_gemm_small(t.Gb, pk_hw2, t.Qf, Mc, Ac, Ec, Ec, stream);
  mask_kernel<<<dim3((Mc*Ac)/256), dim3(256), 0, stream>>>(t.Qf, legal, (float*)d_out, Mc*Ac);
}
